// PDD__74371653698150
// MI455X (gfx1250) — matrix operand order repaired and checked by reading, not compiled, not run on hardware
//
#include <hip/hip_runtime.h>
#include <hip/hip_bf16.h>

typedef __attribute__((ext_vector_type(16))) _Float16 v16h;
typedef __attribute__((ext_vector_type(8)))  _Float16 v8h;
typedef __attribute__((ext_vector_type(4)))  _Float16 v4h;
typedef __attribute__((ext_vector_type(4)))  float    v4f;
typedef __attribute__((ext_vector_type(8)))  float    v8f;

// Problem constants (match reference)
#define BN_   8
#define CC_   256
#define HH_   128
#define WW_   128
#define PP_   (HH_ * WW_)       // 16384 pixels per (b,c) plane
#define CKK_  (CC_ * 9)         // 2304
#define EPS_  1e-5f

#ifndef __has_builtin
#define __has_builtin(x) 0
#endif

#if __has_builtin(__builtin_amdgcn_global_load_async_to_lds_b128)
#define USE_ASYNC_LDS 1
// Parameter types per hipcc diagnostic: param0 is
//   'int __attribute__((vector_size(16))) __device__ *'  (AS1 = HIP __device__)
typedef int v4i_ __attribute__((vector_size(16)));
typedef __attribute__((address_space(1))) v4i_* gas_v4i;
typedef __attribute__((address_space(3))) v4i_* las_v4i;
#else
#define USE_ASYNC_LDS 0
#endif

static __device__ __forceinline__ void wait_async_lds() {
#if USE_ASYNC_LDS
#if __has_builtin(__builtin_amdgcn_s_wait_asynccnt)
    __builtin_amdgcn_s_wait_asynccnt(0);
#else
    asm volatile("s_wait_asynccnt 0" ::: "memory");
#endif
#endif
}

static __device__ __forceinline__ float sigmoidf_(float x) {
    return 1.0f / (1.0f + __expf(-x));
}

// ---------------------------------------------------------------------------
// 0) fp32 -> fp16 conversion (scalar for small weights, x4 for tensors)
// ---------------------------------------------------------------------------
__global__ __launch_bounds__(256) void cvt_f16_kernel(const float* __restrict__ src,
                                                      _Float16* __restrict__ dst, int n) {
    int i = blockIdx.x * 256 + threadIdx.x;
    if (i < n) dst[i] = (_Float16)src[i];
}

__global__ __launch_bounds__(256) void cvt4_f16_kernel(const float* __restrict__ src,
                                                       _Float16* __restrict__ dst, int n4) {
    int i = blockIdx.x * 256 + threadIdx.x;
    if (i < n4) {
        v4f f = ((const v4f*)src)[i];
        v4h h;
        h[0] = (_Float16)f[0]; h[1] = (_Float16)f[1];
        h[2] = (_Float16)f[2]; h[3] = (_Float16)f[3];
        ((v4h*)dst)[i] = h;
    }
}

// ---------------------------------------------------------------------------
// 1) Global average pool: one block per (b,c)
// ---------------------------------------------------------------------------
__global__ __launch_bounds__(256) void pool_kernel(const float* __restrict__ x,
                                                   float* __restrict__ pooled) {
    __shared__ float red[256];
    int bc = blockIdx.x;
    int tid = threadIdx.x;
    const float* xp = x + (size_t)bc * PP_;
    float s = 0.f;
    for (int i = tid; i < PP_; i += 256) s += xp[i];
    red[tid] = s;
    __syncthreads();
    for (int st = 128; st > 0; st >>= 1) {
        if (tid < st) red[tid] += red[tid + st];
        __syncthreads();
    }
    if (tid == 0) pooled[bc] = red[0] * (1.0f / (float)PP_);
}

// ---------------------------------------------------------------------------
// 2) Dynamic kernel generation: a = sigmoid(BN(pooled @ ka_w.T + ka_b))
// ---------------------------------------------------------------------------
__global__ __launch_bounds__(256) void ka_kernel(const float* __restrict__ pooled,
                                                 const float* __restrict__ ka_w,
                                                 const float* __restrict__ ka_b,
                                                 const float* __restrict__ bn_g,
                                                 const float* __restrict__ bn_b,
                                                 const float* __restrict__ bn_m,
                                                 const float* __restrict__ bn_v,
                                                 float* __restrict__ dynw) {
    __shared__ float pl[CC_];
    int tid = threadIdx.x;
    int bi = blockIdx.y;
    pl[tid] = pooled[bi * CC_ + tid];
    __syncthreads();
    int o = blockIdx.x * 256 + tid;      // 0 .. CKK_-1
    const float* wrow = ka_w + (size_t)o * CC_;
    float s = 0.f;
    #pragma unroll 8
    for (int c = 0; c < CC_; ++c) s += pl[c] * wrow[c];
    s += ka_b[o];
    s = (s - bn_m[o]) * (bn_g[o] * rsqrtf(bn_v[o] + EPS_)) + bn_b[o];
    dynw[(size_t)bi * CKK_ + o] = sigmoidf_(s);
}

// ---------------------------------------------------------------------------
// 3) Depthwise 3x3 with reflect padding, static weights (C,9); OutT = f16/f32
// ---------------------------------------------------------------------------
static __device__ __forceinline__ int refl_(int i) {
    return i < 0 ? -i : (i > (HH_ - 1) ? 2 * (HH_ - 1) - i : i);
}

template <typename OutT>
__global__ __launch_bounds__(256) void dw_reflect_kernel(const float* __restrict__ in,
                                                         const float* __restrict__ wd,
                                                         const float* __restrict__ bias,
                                                         OutT* __restrict__ out) {
    size_t idx = (size_t)blockIdx.x * 256 + threadIdx.x;   // b*C*H*W threads
    int hw = (int)(idx & (PP_ - 1));
    int bc = (int)(idx >> 14);
    int c  = bc & (CC_ - 1);
    int h = hw >> 7, w = hw & (WW_ - 1);
    const float* ip = in + (size_t)bc * PP_;
    const float* wk = wd + c * 9;
    float acc = 0.f;
    #pragma unroll
    for (int i = 0; i < 3; ++i) {
        int hh = refl_(h - 1 + i);
        #pragma unroll
        for (int j = 0; j < 3; ++j) {
            int ww = refl_(w - 1 + j);
            acc += wk[i * 3 + j] * ip[hh * WW_ + ww];
        }
    }
    out[idx] = (OutT)(acc + bias[c]);
}

// ---------------------------------------------------------------------------
// 4) 1x1 conv as GEMM with V_WMMA_F32_16X16X32_F16; f16 inputs/weights.
//    Weight panel staged once into LDS (async global->LDS when available);
//    input tile staged transposed [pixel][k] so both fragments are
//    contiguous-K per lane -> ds_load_b128 reads.
//    ACT: 0=none, 1=relu, 2=sigmoid.  OutT: float or _Float16.
// ---------------------------------------------------------------------------
template <int COUT, int CIN, int MW, int ACT, typename OutT>
__global__ __launch_bounds__(256) void gemm1x1_kernel(const _Float16* __restrict__ in,
                                                      const _Float16* __restrict__ wgt,
                                                      const float* __restrict__ bias,
                                                      OutT* __restrict__ out) {
    constexpr int NW    = 8 / MW;
    constexpr int NPIX  = NW * 16;
    constexpr int MROWS = MW * 16;

    __shared__ _Float16 sW[MROWS][CIN + 8];     // [co][k] whole panel, 16B-aligned rows
    __shared__ _Float16 sInT[NPIX][40];         // [pixel][k-chunk] transposed tile

    const int tid  = threadIdx.x;
    const int lane = tid & 31;
    const int wave = tid >> 5;
    const int mt = wave % MW;
    const int nt = wave / MW;
    const int bb = blockIdx.z;
    const int pixBase  = blockIdx.x * NPIX;
    const int coutBase = blockIdx.y * MROWS;

    const _Float16* inB = in + (size_t)bb * CIN * PP_;

    // ---- stage full weight panel once: MROWS x CIN halves in 16B chunks ----
    constexpr int WCH = MROWS * CIN / 8;
#if USE_ASYNC_LDS
    for (int i = tid; i < WCH; i += 256) {
        int mi = i / (CIN / 8);
        int k8 = (i % (CIN / 8)) * 8;
        const _Float16* gp = wgt + (size_t)(coutBase + mi) * CIN + k8;
        _Float16* lp = &sW[mi][k8];
        __builtin_amdgcn_global_load_async_to_lds_b128((gas_v4i)gp, (las_v4i)lp, 0, 0);
    }
    wait_async_lds();
#else
    for (int i = tid; i < WCH; i += 256) {
        int mi = i / (CIN / 8);
        int k8 = (i % (CIN / 8)) * 8;
        *(v8h*)&sW[mi][k8] = *(const v8h*)(wgt + (size_t)(coutBase + mi) * CIN + k8);
    }
#endif
    __syncthreads();

    v8f acc = {};

    for (int kc = 0; kc < CIN; kc += 32) {
        // ---- stage input tile transposed: sInT[pix][k] ----
        for (int i = tid; i < (32 * NPIX) / 4; i += 256) {
            int ci = i / (NPIX / 4);
            int p4 = (i % (NPIX / 4)) * 4;
            const _Float16* gp = inB + (size_t)(kc + ci) * PP_ + pixBase + p4;
            v4h tv = *(const v4h*)gp;
            sInT[p4 + 0][ci] = tv[0];
            sInT[p4 + 1][ci] = tv[1];
            sInT[p4 + 2][ci] = tv[2];
            sInT[p4 + 3][ci] = tv[3];
            if (kc + 32 < CIN) __builtin_prefetch(gp + 32 * PP_, 0, 1);
        }
        __syncthreads();

        // ---- fragments: contiguous-K vector reads (CDNA5 ISA §7.12.2)
        const int row = lane & 15;
        const int kbA = ((lane >> 4) << 3);   // A: hi-half lanes hold K base +8
        const int kbB = ((lane >> 4) << 3);   // B: hi-half lanes hold K base +8
        const v8h a0 = *(const v8h*)&sW[mt * 16 + row][kc + kbA];
        const v8h a1 = *(const v8h*)&sW[mt * 16 + row][kc + kbA + 16];
        const v8h b0 = *(const v8h*)&sInT[nt * 16 + row][kbB];
        const v8h b1 = *(const v8h*)&sInT[nt * 16 + row][kbB + 16];
        v16h a = __builtin_shufflevector(a0, a1, 0,1,2,3,4,5,6,7,8,9,10,11,12,13,14,15);
        v16h b = __builtin_shufflevector(b0, b1, 0,1,2,3,4,5,6,7,8,9,10,11,12,13,14,15);

        acc = __builtin_amdgcn_wmma_f32_16x16x32_f16(
            /*neg_a=*/false, a, /*neg_b=*/false, b,
            /*c_mod=*/(short)0, acc, /*reuse_a=*/false, /*reuse_b=*/false);
        __syncthreads();
    }

    // ---- epilogue: C/D layout — lane = column N, VGPR r -> row r (+8 hi lanes)
    const int n     = lane & 15;
    const int mhalf = (lane >> 4) << 3;
    const int pix   = pixBase + nt * 16 + n;
    #pragma unroll
    for (int r = 0; r < 8; ++r) {
        int co = coutBase + mt * 16 + mhalf + r;
        float v = acc[r] + bias[co];
        if (ACT == 1) v = fmaxf(v, 0.f);
        else if (ACT == 2) v = sigmoidf_(v);
        out[((size_t)bb * COUT + co) * PP_ + pix] = (OutT)v;
    }
}

// ---------------------------------------------------------------------------
// 5) Final: xb = t*v; xd = x - xb (zero-pad); y = dyndw3x3(xd) + bias + xb
// ---------------------------------------------------------------------------
__global__ __launch_bounds__(256) void final_kernel(const float* __restrict__ x,
                                                    const float* __restrict__ t,
                                                    const float* __restrict__ v,
                                                    const float* __restrict__ dynw,
                                                    const float* __restrict__ bias,
                                                    float* __restrict__ out) {
    __shared__ float xds[18][20];
    __shared__ float xbs[18][20];
    const int bc = blockIdx.z;
    const int c  = bc & (CC_ - 1);
    const int h0 = blockIdx.y * 16;
    const int w0 = blockIdx.x * 16;
    const size_t base = (size_t)bc * PP_;

    for (int i = threadIdx.x; i < 18 * 18; i += 256) {
        int r = i / 18, cc = i % 18;
        int gh = h0 + r - 1, gw = w0 + cc - 1;
        float xd = 0.f, xb = 0.f;
        if (gh >= 0 && gh < HH_ && gw >= 0 && gw < WW_) {
            size_t off = base + gh * WW_ + gw;
            xb = t[off] * v[off];
            xd = x[off] - xb;
        }
        xds[r][cc] = xd;
        xbs[r][cc] = xb;
    }
    __syncthreads();

    const int ty = threadIdx.x >> 4;
    const int tx = threadIdx.x & 15;
    const float* wk = dynw + (size_t)bc * 9;
    float acc = 0.f;
    #pragma unroll
    for (int i = 0; i < 3; ++i)
        #pragma unroll
        for (int j = 0; j < 3; ++j)
            acc += wk[i * 3 + j] * xds[ty + i][tx + j];

    out[base + (h0 + ty) * WW_ + (w0 + tx)] = acc + bias[c] + xbs[ty + 1][tx + 1];
}

// ---------------------------------------------------------------------------
// Host-side orchestration
// ---------------------------------------------------------------------------
extern "C" void kernel_launch(void* const* d_in, const int* in_sizes, int n_in,
                              void* d_out, int out_size, void* d_ws, size_t ws_size,
                              hipStream_t stream) {
    const float* x       = (const float*)d_in[0];
    const float* bias    = (const float*)d_in[1];
    const float* ka_w    = (const float*)d_in[2];
    const float* ka_b    = (const float*)d_in[3];
    const float* ka_bn_g = (const float*)d_in[4];
    const float* ka_bn_b = (const float*)d_in[5];
    const float* ka_bn_m = (const float*)d_in[6];
    const float* ka_bn_v = (const float*)d_in[7];
    const float* wv_w1   = (const float*)d_in[8];
    const float* wv_b1   = (const float*)d_in[9];
    const float* wv_dw   = (const float*)d_in[10];
    const float* wv_dwb  = (const float*)d_in[11];
    const float* td_dw   = (const float*)d_in[12];
    const float* td_dwb  = (const float*)d_in[13];
    const float* td_w1   = (const float*)d_in[14];
    const float* td_b1   = (const float*)d_in[15];
    const float* td_w2   = (const float*)d_in[16];
    const float* td_b2   = (const float*)d_in[17];
    float* out = (float*)d_out;

    // Workspace layout (byte-based, 256B aligned regions)
    char* wsb = (char*)d_ws;
    auto alloc = [&](size_t bytes) -> char* {
        char* p = wsb;
        wsb += (bytes + 255) & ~(size_t)255;
        return p;
    };
    const size_t NBIG = (size_t)BN_ * CC_ * PP_;                 // 33,554,432 elems
    float*    pooled = (float*)   alloc((size_t)BN_ * CC_ * 4);
    float*    dynw   = (float*)   alloc((size_t)BN_ * CKK_ * 4);
    _Float16* wf_wv  = (_Float16*)alloc((size_t)CC_ * CC_ * 2);
    _Float16* wf_t1  = (_Float16*)alloc((size_t)(CC_ / 8) * CC_ * 2);
    _Float16* wf_t2  = (_Float16*)alloc((size_t)CC_ * (CC_ / 8) * 2);
    _Float16* xh     = (_Float16*)alloc(NBIG * 2);
    _Float16* t1h    = (_Float16*)alloc(NBIG * 2);
    _Float16* hidden = (_Float16*)alloc((NBIG / 8) * 2);
    float*    tb     = (float*)   alloc(NBIG * 4);
    float*    ub     = (float*)   alloc(NBIG * 4);
    float*    vb     = (float*)   alloc(NBIG * 4);

    // 0) conversions to f16
    cvt_f16_kernel<<<(CC_ * CC_ + 255) / 256, 256, 0, stream>>>(wv_w1, wf_wv, CC_ * CC_);
    cvt_f16_kernel<<<((CC_ / 8) * CC_ + 255) / 256, 256, 0, stream>>>(td_w1, wf_t1, (CC_ / 8) * CC_);
    cvt_f16_kernel<<<(CC_ * (CC_ / 8) + 255) / 256, 256, 0, stream>>>(td_w2, wf_t2, CC_ * (CC_ / 8));
    cvt4_f16_kernel<<<(int)(NBIG / 4 / 256), 256, 0, stream>>>(x, xh, (int)(NBIG / 4));

    // 1) pool + 2) dynamic kernel generation
    pool_kernel<<<BN_ * CC_, 256, 0, stream>>>(x, pooled);
    ka_kernel<<<dim3(CKK_ / 256, BN_), 256, 0, stream>>>(pooled, ka_w, ka_b,
                                                         ka_bn_g, ka_bn_b, ka_bn_m, ka_bn_v, dynw);

    const int NDW = (int)(NBIG / 256);

    // 3) t-path: t1 = dw_reflect(x) [f16 out] -> hidden = relu(1x1 256->32) -> t = sigmoid(1x1 32->256)
    dw_reflect_kernel<_Float16><<<NDW, 256, 0, stream>>>(x, td_dw, td_dwb, t1h);
    gemm1x1_kernel<32, 256, 2, 1, _Float16>
        <<<dim3(PP_ / 64, 1, BN_), 256, 0, stream>>>(t1h, wf_t1, td_b1, hidden);
    gemm1x1_kernel<256, 32, 4, 2, float>
        <<<dim3(PP_ / 32, CC_ / 64, BN_), 256, 0, stream>>>(hidden, wf_t2, td_b2, tb);

    // 4) v-path: u = 1x1 256->256 (xh), then v = dw_reflect(u)
    gemm1x1_kernel<256, 256, 4, 0, float>
        <<<dim3(PP_ / 32, CC_ / 64, BN_), 256, 0, stream>>>(xh, wf_wv, wv_b1, ub);
    dw_reflect_kernel<float><<<NDW, 256, 0, stream>>>(ub, wv_dw, wv_dwb, vb);

    // 5) fused dynamic depthwise + residual
    final_kernel<<<dim3(WW_ / 16, HH_ / 16, BN_ * CC_), 256, 0, stream>>>(x, tb, vb, dynw, bias, out);

    (void)in_sizes; (void)n_in; (void)out_size; (void)ws_size;
}
